// Net_F_80204219286024
// MI455X (gfx1250) — matrix operand order repaired and checked by reading, not compiled, not run on hardware
//
#include <hip/hip_runtime.h>
#include <hip/hip_bf16.h>

typedef __attribute__((ext_vector_type(16))) _Float16 v16h;
typedef __attribute__((ext_vector_type(8)))  _Float16 v8h;
typedef __attribute__((ext_vector_type(8)))  float    v8f;

#define HID        256
#define PITCH      264                    // padded k-pitch (f16 elems) -> no LDS bank conflicts
#define ROWS       16                     // points per workgroup
#define NST        7                      // value, 3 first-deriv, 3 second-deriv streams
#define ACT_ELEMS  (NST * ROWS * PITCH)   // f16 elems per activation buffer
#define NU_INV     100.0f
#define NPTS       65536

// Hardware tanh on gfx1250 (V_TANH_F32, TRANS op -> co-executes with WMMA).
__device__ __forceinline__ float hw_tanh(float a) {
#if __has_builtin(__builtin_amdgcn_tanhf)
    return __builtin_amdgcn_tanhf(a);
#else
    return tanhf(a);
#endif
}

// ---------------------------------------------------------------------------
// Pack W1..W5 (256x256 fp32, row-major [k][n]) into f16 transposed [n][k]
// so a WMMA B-fragment (k = kbase + 0..15 of one column n) is one contiguous
// 32-byte load per lane. Also pack the 5 hidden biases into fp32 scratch.
// ---------------------------------------------------------------------------
__global__ void pack_params_kernel(const float* __restrict__ W1, const float* __restrict__ W2,
                                   const float* __restrict__ W3, const float* __restrict__ W4,
                                   const float* __restrict__ W5,
                                   const float* __restrict__ b1, const float* __restrict__ b2,
                                   const float* __restrict__ b3, const float* __restrict__ b4,
                                   const float* __restrict__ b5,
                                   _Float16* __restrict__ dstW, float* __restrict__ dstB)
{
    int g = blockIdx.x * blockDim.x + threadIdx.x;
    if (g < 5 * 65536) {
        int layer = g >> 16;
        int e = g & 65535;
        int n = e >> 8;          // output column
        int k = e & 255;         // input index
        const float* W = (layer == 0) ? W1 : (layer == 1) ? W2 :
                         (layer == 2) ? W3 : (layer == 3) ? W4 : W5;
        dstW[g] = (_Float16)W[k * 256 + n];
    } else {
        int i = g - 5 * 65536;
        if (i < 5 * 256) {
            int layer = i >> 8;
            int n = i & 255;
            const float* B = (layer == 0) ? b1 : (layer == 1) ? b2 :
                             (layer == 2) ? b3 : (layer == 3) ? b4 : b5;
            dstB[i] = B[n];
        }
    }
}

// ---------------------------------------------------------------------------
// Main kernel: second-order forward-mode jet propagation through the MLP.
// One workgroup (8 wave32) handles 16 collocation points; activations for the
// 7 jet streams live in LDS as a (7*16) x 256 f16 matrix (double buffered).
// Each wave owns 2 output n-tiles per layer with BOTH accumulator sets live
// (14 x v8f), so every A fragment is loaded from LDS exactly once and feeds
// two back-to-back WMMAs (same A operand, different B).
// ---------------------------------------------------------------------------
__global__ __launch_bounds__(256)
void pinn_hessian_kernel(const float* __restrict__ x, const float* __restrict__ y,
                         const float* __restrict__ t,
                         const float* __restrict__ W0, const float* __restrict__ b0,
                         const _Float16* __restrict__ Wt,   // 5 x [256][256] f16, [n][k]
                         const float* __restrict__ Bs,      // 5 x [256] fp32
                         const float* __restrict__ W6,      // [256] fp32
                         float* __restrict__ out)
{
    extern __shared__ __align__(16) _Float16 smem[];
    _Float16* actA = smem;
    _Float16* actB = smem + ACT_ELEMS;
    float*    red  = (float*)(smem + 2 * ACT_ELEMS);   // 16x16 reduction scratch

    const int tid  = threadIdx.x;
    const int lane = tid & 31;
    const int wave = tid >> 5;
    const int row0 = blockIdx.x * ROWS;

    // ---- Layer 1: 3 -> 256, analytic jets (a' = W0[d,:], a'' = 0) ----
    {
        const int r  = tid >> 4;          // point row 0..15
        const int c0 = (tid & 15) << 4;   // 16-col chunk
        const float px = x[row0 + r];
        const float py = y[row0 + r];
        const float pt = t[row0 + r];
        #pragma unroll 4
        for (int j = 0; j < 16; ++j) {
            const int n = c0 + j;
            const float w0 = W0[n], w1 = W0[256 + n], w2 = W0[512 + n];
            const float a  = px * w0 + py * w1 + pt * w2 + b0[n];
            const float s  = hw_tanh(a);
            const float tt = 1.0f - s * s;
            const float c2 = -2.0f * s * tt;
            _Float16* base = actA + r * PITCH + n;     // [stream][m][k] layout
            base[0 * ROWS * PITCH] = (_Float16)s;
            base[1 * ROWS * PITCH] = (_Float16)(tt * w0);
            base[2 * ROWS * PITCH] = (_Float16)(tt * w1);
            base[3 * ROWS * PITCH] = (_Float16)(tt * w2);
            base[4 * ROWS * PITCH] = (_Float16)(c2 * w0 * w0);
            base[5 * ROWS * PITCH] = (_Float16)(c2 * w1 * w1);
            base[6 * ROWS * PITCH] = (_Float16)(c2 * w2 * w2);
        }
    }

    // ---- Hidden layers 2..6: five 256x256 WMMA matmuls over 7 streams ----
    _Float16* cur = actA;
    _Float16* nxt = actB;

    const int m_of  = lane & 15;          // A row / D col-within-tile
    const int ksel  = (lane >> 4) << 4;   // A-frag k sub-offset (contiguous k = 16h + s, matches B)
    const int kbsel = (lane >> 4) << 4;   // B-frag k sub-offset
    const int ncol0 = (wave << 5) + m_of;        // n-tile 2*wave
    const int ncol1 = (wave << 5) + 16 + m_of;   // n-tile 2*wave+1

    for (int layer = 0; layer < 5; ++layer) {
        __syncthreads();
        const _Float16* W   = Wt + layer * 65536;
        const float*   bias = Bs + layer * 256;

        v8f acc0[NST] = {};
        v8f acc1[NST] = {};
        #pragma unroll
        for (int kt = 0; kt < 8; ++kt) {
            const int kb = kt << 5;
            // B fragments for both n-tiles: contiguous 32B per lane
            const v16h bf0 = *(const v16h*)(W + ncol0 * 256 + kb + kbsel);
            const v16h bf1 = *(const v16h*)(W + ncol1 * 256 + kb + kbsel);
            #pragma unroll
            for (int st = 0; st < NST; ++st) {
                // A fragment: row m_of, two contiguous 16B half-chunks (once!)
                const _Float16* ap = cur + (st * ROWS + m_of) * PITCH + kb + ksel;
                const v8h lo = *(const v8h*)ap;
                const v8h hi = *(const v8h*)(ap + 8);
                const v16h af = __builtin_shufflevector(lo, hi,
                    0,1,2,3,4,5,6,7,8,9,10,11,12,13,14,15);
                acc0[st] = __builtin_amdgcn_wmma_f32_16x16x32_f16(
                    false, af, false, bf0, (short)0, acc0[st], false, false);
                acc1[st] = __builtin_amdgcn_wmma_f32_16x16x32_f16(
                    false, af, false, bf1, (short)0, acc1[st], false, false);
            }
        }

        // ---- elementwise tanh jet update (fp32), store f16 for next layer
        #pragma unroll
        for (int half = 0; half < 2; ++half) {
            v8f* acc = half ? acc1 : acc0;
            const int ncol = half ? ncol1 : ncol0;
            const float bv = bias[ncol];
            #pragma unroll
            for (int r2 = 0; r2 < 8; ++r2) {
                const float a0 = acc[0][r2] + bv;
                const float s  = hw_tanh(a0);
                const float tt = 1.0f - s * s;
                const float a1 = acc[1][r2], a2 = acc[2][r2], a3 = acc[3][r2];
                const int   m  = r2 + ((lane >> 4) << 3);      // ISA f32 C/D layout
                _Float16* dbase = nxt + m * PITCH + ncol;
                dbase[0 * ROWS * PITCH] = (_Float16)s;
                dbase[1 * ROWS * PITCH] = (_Float16)(tt * a1);
                dbase[2 * ROWS * PITCH] = (_Float16)(tt * a2);
                dbase[3 * ROWS * PITCH] = (_Float16)(tt * a3);
                dbase[4 * ROWS * PITCH] = (_Float16)(tt * (acc[4][r2] - 2.0f * s * a1 * a1));
                dbase[5 * ROWS * PITCH] = (_Float16)(tt * (acc[5][r2] - 2.0f * s * a2 * a2));
                dbase[6 * ROWS * PITCH] = (_Float16)(tt * (acc[6][r2] - 2.0f * s * a3 * a3));
            }
        }
        _Float16* tmp = cur; cur = nxt; nxt = tmp;
    }
    __syncthreads();

    // ---- Output layer: f = W6^T (h''_x + h''_y - 100 * h''_t) per point ----
    {
        const int r = tid >> 4;
        const int p = tid & 15;
        float partial = 0.0f;
        #pragma unroll 4
        for (int j = 0; j < 16; ++j) {
            const int k = (p << 4) + j;
            const float w  = W6[k];
            const float hx = (float)cur[(4 * ROWS + r) * PITCH + k];
            const float hy = (float)cur[(5 * ROWS + r) * PITCH + k];
            const float ht = (float)cur[(6 * ROWS + r) * PITCH + k];
            partial += w * (hx + hy - NU_INV * ht);
        }
        red[(r << 4) + p] = partial;
    }
    __syncthreads();
    if (tid < 16) {
        float sum = 0.0f;
        #pragma unroll
        for (int j = 0; j < 16; ++j) sum += red[(tid << 4) + j];
        out[row0 + tid] = sum;
    }
}

// ---------------------------------------------------------------------------
extern "C" void kernel_launch(void* const* d_in, const int* in_sizes, int n_in,
                              void* d_out, int out_size, void* d_ws, size_t ws_size,
                              hipStream_t stream)
{
    // setup_inputs order: x, y, t, then params flattened W0,b0,...,W6,b6
    const float* x  = (const float*)d_in[0];
    const float* y  = (const float*)d_in[1];
    const float* t  = (const float*)d_in[2];
    const float* W0 = (const float*)d_in[3];
    const float* b0 = (const float*)d_in[4];
    const float* W1 = (const float*)d_in[5];
    const float* b1 = (const float*)d_in[6];
    const float* W2 = (const float*)d_in[7];
    const float* b2 = (const float*)d_in[8];
    const float* W3 = (const float*)d_in[9];
    const float* b3 = (const float*)d_in[10];
    const float* W4 = (const float*)d_in[11];
    const float* b4 = (const float*)d_in[12];
    const float* W5 = (const float*)d_in[13];
    const float* b5 = (const float*)d_in[14];
    const float* W6 = (const float*)d_in[15];
    // b6 (d_in[16]) does not appear in any second derivative.

    _Float16* wsW = (_Float16*)d_ws;                                      // 5*65536 f16
    float*    wsB = (float*)((char*)d_ws + 5 * 65536 * sizeof(_Float16)); // 5*256 f32

    pack_params_kernel<<<1285, 256, 0, stream>>>(W1, W2, W3, W4, W5,
                                                 b1, b2, b3, b4, b5, wsW, wsB);

    const size_t shmem = 2 * ACT_ELEMS * sizeof(_Float16) + 16 * 16 * sizeof(float);
    pinn_hessian_kernel<<<NPTS / ROWS, 256, shmem, stream>>>(
        x, y, t, W0, b0, wsW, wsB, W6, (float*)d_out);
}
